// AdvancedMuonAttention_63428077027728
// MI455X (gfx1250) — matrix operand order repaired and checked by reading, not compiled, not run on hardware
//
#include <hip/hip_runtime.h>

// ---------------------------------------------------------------------------
// Types for CDNA5 WMMA (wave32): bf16 A/B fragments (v16bf), f32 C/D (v8f)
// ---------------------------------------------------------------------------
typedef __attribute__((ext_vector_type(16))) __bf16 v16bf;
typedef __attribute__((ext_vector_type(8)))  __bf16 v8bf;
typedef __attribute__((ext_vector_type(8)))  float  v8f;

#define D_MODEL 1024
#define NUM_HEADS 16
#define D_K 64
#define BB 2
#define SS 2048

__device__ __forceinline__ __bf16 f2bf(float f) {
  unsigned u = __builtin_bit_cast(unsigned, f);
  unsigned r = u + 0x7FFFu + ((u >> 16) & 1u);   // round-to-nearest-even
  unsigned short h = (unsigned short)(r >> 16);
  return __builtin_bit_cast(__bf16, h);
}

// Assemble a 16-element bf16 fragment from two contiguous 8-element chunks
// (each chunk is one 16-byte load -> ds_load_b128 / global_load_b128).
__device__ __forceinline__ v16bf ld16(const __bf16* p0, const __bf16* p1) {
  v8bf lo = *(const v8bf*)p0;
  v8bf hi = *(const v8bf*)p1;
  v16bf r;
#pragma unroll
  for (int i = 0; i < 8; ++i) { r[i] = lo[i]; r[i + 8] = hi[i]; }
  return r;
}

__device__ __forceinline__ v8f wmma_bf16(v16bf a, v16bf b, v8f c) {
  return __builtin_amdgcn_wmma_f32_16x16x32_bf16(
      false, a, false, b, (short)0, c, /*reuse_a=*/false, /*reuse_b=*/false);
}
// reuse_b hint: B operand cached for the (identical) next WMMA instruction
__device__ __forceinline__ v8f wmma_bf16_rb(v16bf a, v16bf b, v8f c) {
  return __builtin_amdgcn_wmma_f32_16x16x32_bf16(
      false, a, false, b, (short)0, c, /*reuse_a=*/false, /*reuse_b=*/true);
}

// Generic LDS pointer -> 32-bit LDS byte offset (low 32 bits of shared aperture)
__device__ __forceinline__ unsigned lds_off(const void* p) {
  return (unsigned)(unsigned long long)p;
}

// gfx1250 async copy global -> LDS (16B), tracked by ASYNCcnt, no VGPR traffic.
__device__ __forceinline__ void async_ld_b128(unsigned lds, const void* gptr) {
  asm volatile("global_load_async_to_lds_b128 %0, %1, off"
               :: "v"(lds), "v"(gptr) : "memory");
}
__device__ __forceinline__ void wait_async_0() {
  asm volatile("s_wait_asynccnt 0x0" ::: "memory");
}
__device__ __forceinline__ void wait_async_4() {
  asm volatile("s_wait_asynccnt 0x4" ::: "memory");
}

// ---------------------------------------------------------------------------
// Kernel 1a: fp32 -> bf16 conversion (activations), 4 elems/thread
// ---------------------------------------------------------------------------
__global__ void cvt_f32_bf16(const float* __restrict__ in, __bf16* __restrict__ out, int n) {
  int i = (blockIdx.x * blockDim.x + threadIdx.x) * 4;
  if (i < n) {
    float4 v = *(const float4*)(in + i);
    out[i]     = f2bf(v.x);
    out[i + 1] = f2bf(v.y);
    out[i + 2] = f2bf(v.z);
    out[i + 3] = f2bf(v.w);
  }
}

// ---------------------------------------------------------------------------
// Kernel 1b: fp32 [K,N] -> bf16 transposed [N,K] (weights), tiled via LDS.
// block (32,8); grid (N/32, K/32)
// ---------------------------------------------------------------------------
__global__ void cvt_transpose(const float* __restrict__ in, __bf16* __restrict__ out,
                              int K, int N) {
  __shared__ float tile[32][33];
  int kb = blockIdx.y * 32, nb = blockIdx.x * 32;
  for (int r = threadIdx.y; r < 32; r += 8)
    tile[r][threadIdx.x] = in[(size_t)(kb + r) * N + nb + threadIdx.x];
  __syncthreads();
  for (int r = threadIdx.y; r < 32; r += 8)
    out[(size_t)(nb + r) * K + kb + threadIdx.x] = f2bf(tile[threadIdx.x][r]);
}

// ---------------------------------------------------------------------------
// Kernel 1c: pack int mask -> bit image via wave32 ballot (bit l = lane l)
// ---------------------------------------------------------------------------
__global__ void mask_pack(const int* __restrict__ mask, unsigned* __restrict__ bits) {
  int i = blockIdx.x * blockDim.x + threadIdx.x;
  unsigned long long bal = __ballot(mask[i] != 0);
  if ((threadIdx.x & 31) == 0) bits[i >> 5] = (unsigned)bal;
}

// ---------------------------------------------------------------------------
// Kernel 2: tiled bf16 GEMM, C = act(A[M,K] @ B + bias), B given TRANSPOSED [N,K].
// Block 256 threads = 8 waves; tile 128(M) x 128(N); K step 32, double-buffered
// async global->LDS staging. Waves 4(M) x 2(N); wave tile 32x64 = 8 WMMA/step.
// mode: 0 = bf16 out, 1 = bf16 + tanh, 2 = f32 out, 3 = bf16 out in [B,H,D,S]
// ---------------------------------------------------------------------------
__global__ void gemm_bf16(const __bf16* __restrict__ A, const __bf16* __restrict__ BT,
                          const float* __restrict__ bias, void* __restrict__ Cout,
                          int M, int N, int K, int mode) {
  __shared__ __bf16 aT[2][128][32];   // [buf][m][k]   8KB each
  __shared__ __bf16 bT[2][128][32];   // [buf][n][k]   8KB each

  const int t = threadIdx.x;
  const int wid = t >> 5, lane = t & 31;
  const int lanelo = lane & 15, hl = lane >> 4;
  const int wm = wid & 3, wn = wid >> 2;
  const int m0 = blockIdx.y * 128;
  const int n0 = blockIdx.x * 128;
  const int rowW = wm * 32, colW = wn * 64;

  v8f acc[2][4];
#pragma unroll
  for (int g = 0; g < 2; ++g)
#pragma unroll
    for (int i = 0; i < 4; ++i) acc[g][i] = v8f{};

  // issue one K-stage: 1024 x 16B async loads per tile pair (4 per thread)
  auto stage = [&](int k0, int bufi) {
#pragma unroll
    for (int j = 0; j < 2; ++j) {
      int linear = t + j * 256;          // 0..511
      int row = linear >> 2;             // 0..127
      int kc = (linear & 3) * 8;         // 0,8,16,24
      async_ld_b128(lds_off(&aT[bufi][row][kc]),
                    A + (size_t)(m0 + row) * K + k0 + kc);
      int bn = n0 + row; if (bn > N - 1) bn = N - 1;   // clamp (N < 128 tiles)
      async_ld_b128(lds_off(&bT[bufi][row][kc]),
                    BT + (size_t)bn * K + k0 + kc);
    }
  };

  const int nstages = K / 32;
  stage(0, 0);
  for (int s = 0; s < nstages; ++s) {
    const int cur = s & 1;
    __syncthreads();                        // prior reads of buf cur^1 done
    if (s + 1 < nstages) { stage((s + 1) * 32, cur ^ 1); wait_async_4(); }
    else                 { wait_async_0(); }
    __syncthreads();                        // stage s landed for all waves

    v16bf aF0 = ld16(&aT[cur][rowW + lanelo][hl * 8],
                     &aT[cur][rowW + lanelo][hl * 8 + 16]);
    v16bf aF1 = ld16(&aT[cur][rowW + 16 + lanelo][hl * 8],
                     &aT[cur][rowW + 16 + lanelo][hl * 8 + 16]);
#pragma unroll
    for (int tt = 0; tt < 4; ++tt) {
      int col = colW + tt * 16 + lanelo;
      v16bf bF = ld16(&bT[cur][col][hl * 8], &bT[cur][col][hl * 8 + 16]);
      acc[0][tt] = wmma_bf16_rb(aF0, bF, acc[0][tt]);  // hint: reuse B next
      acc[1][tt] = wmma_bf16(aF1, bF, acc[1][tt]);
    }
  }

  // Epilogue: C layout per tile -> lane holds (row = r + hl*8, col = lanelo)
#pragma unroll
  for (int g = 0; g < 2; ++g) {
#pragma unroll
    for (int tt = 0; tt < 4; ++tt) {
      int gn = n0 + colW + tt * 16 + lanelo;
      if (gn >= N) continue;
      float bvv = bias[gn];
#pragma unroll
      for (int r = 0; r < 8; ++r) {
        int gm = m0 + rowW + g * 16 + r + hl * 8;
        float val = acc[g][tt][r] + bvv;
        if (mode == 1) val = tanhf(val);
        if (mode == 2) {
          ((float*)Cout)[(size_t)gm * N + gn] = val;
        } else if (mode == 3) {
          // V layout for attention: gm = b*S+s, gn = h*64+d -> [B,H,D_K,S]
          int bb = gm >> 11, sl = gm & (SS - 1);
          int hh = gn >> 6,  dd = gn & (D_K - 1);
          ((__bf16*)Cout)[(((size_t)bb * NUM_HEADS + hh) * D_K + dd) * SS + sl] = f2bf(val);
        } else {
          ((__bf16*)Cout)[(size_t)gm * N + gn] = f2bf(val);
        }
      }
    }
  }
}

// ---------------------------------------------------------------------------
// Kernel 3: flash attention per head (bf16 WMMA, online softmax).
// grid = (S/128, H, B); block = 256 (8 waves). Each wave: one 16-row q tile,
// 64 keys per iteration (16 WMMA / iter), double-buffered async kf/v staging.
// qf/kf layout: [B,S,H,D_K]; vt layout: [B,H,D_K,S]; mask as packed bits.
// ---------------------------------------------------------------------------
__global__ void flash_attn(const __bf16* __restrict__ qf, const __bf16* __restrict__ kf,
                           const __bf16* __restrict__ vt, const unsigned* __restrict__ mbits,
                           const float* __restrict__ temp, __bf16* __restrict__ ctx) {
  __shared__ __bf16 kfT[2][64][64];   // [buf][key][d]  16KB
  __shared__ __bf16 vS[2][64][64];    // [buf][d][key]  16KB
  __shared__ __bf16 pS[8][16][64];    // per-wave P staging [qrow][key] 16KB

  const int b = blockIdx.z, h = blockIdx.y;
  const int t = threadIdx.x;
  const int wid = t >> 5, lane = t & 31;
  const int lanelo = lane & 15, hl = lane >> 4;
  const int qbase = blockIdx.x * 128 + wid * 16;

  const float scale = 0.125f / temp[h];   // 1/sqrt(D_K) / temperature[h]

  // Q fragments for this wave's 16 rows (K = d, two 32-chunks), from global.
  const size_t qrowbase = (((size_t)b * SS + qbase + lanelo) * NUM_HEADS + h) * D_K;
  v16bf aq0 = ld16(qf + qrowbase + hl * 8,      qf + qrowbase + hl * 8 + 16);
  v16bf aq1 = ld16(qf + qrowbase + 32 + hl * 8, qf + qrowbase + 32 + hl * 8 + 16);

  const size_t kfhead = ((size_t)b * SS) * D_MODEL + (size_t)h * D_K;       // + s*1024 + d
  const size_t vthead = (((size_t)b * NUM_HEADS + h) * D_K) * SS;           // + d*S + s

  v8f acc[4];
#pragma unroll
  for (int i = 0; i < 4; ++i) acc[i] = v8f{};
  float rowm[8], rowl[8];
#pragma unroll
  for (int r = 0; r < 8; ++r) { rowm[r] = -3.0e38f; rowl[r] = 0.0f; }

  // stage one 64-key block: kf tile [key][d] + v tile [d][key]; 4 asyncs/thread
  auto stage = [&](int kb, int bufi) {
#pragma unroll
    for (int j = 0; j < 2; ++j) {
      int linear = t + j * 256;                 // 0..511
      int key = linear >> 3, dc = (linear & 7) * 8;
      async_ld_b128(lds_off(&kfT[bufi][key][dc]),
                    kf + kfhead + (size_t)(kb + key) * D_MODEL + dc);
      int d = linear >> 3, kc = (linear & 7) * 8;
      async_ld_b128(lds_off(&vS[bufi][d][kc]),
                    vt + vthead + (size_t)d * SS + kb + kc);
    }
  };

  const int nstages = SS / 64;
  stage(0, 0);
  for (int s = 0; s < nstages; ++s) {
    const int cur = s & 1;
    const int kb = s * 64;
    __syncthreads();                        // prior reads of buf cur^1 done
    if (s + 1 < nstages) { stage(kb + 64, cur ^ 1); wait_async_4(); }
    else                 { wait_async_0(); }
    __syncthreads();                        // stage s landed for all waves

    // scores: S[16,64] = qf[16,64] @ kf^T  (4 key tiles x 2 WMMAs)
    v8f sc[4];
#pragma unroll
    for (int kt = 0; kt < 4; ++kt) {
      int keyc = kt * 16 + lanelo;
      v16bf b0 = ld16(&kfT[cur][keyc][hl * 8],      &kfT[cur][keyc][hl * 8 + 16]);
      v16bf b1 = ld16(&kfT[cur][keyc][32 + hl * 8], &kfT[cur][keyc][32 + hl * 8 + 16]);
      v8f z = v8f{};
      sc[kt] = wmma_bf16(aq0, b0, z);
      sc[kt] = wmma_bf16(aq1, b1, sc[kt]);
    }

    // online softmax over the 64-key block; C layout: row = r + hl*8, col = lanelo
#pragma unroll
    for (int r = 0; r < 8; ++r) {
      int mrow = qbase + r + hl * 8;
      unsigned w0 = mbits[(size_t)mrow * (SS / 32) + (kb >> 5)];
      unsigned w1 = mbits[(size_t)mrow * (SS / 32) + (kb >> 5) + 1];
      float a0 = ((w0 >> lanelo) & 1u)        ? sc[0][r] * scale : -1e9f;
      float a1 = ((w0 >> (lanelo + 16)) & 1u) ? sc[1][r] * scale : -1e9f;
      float a2 = ((w1 >> lanelo) & 1u)        ? sc[2][r] * scale : -1e9f;
      float a3 = ((w1 >> (lanelo + 16)) & 1u) ? sc[3][r] * scale : -1e9f;
      float mx = fmaxf(fmaxf(a0, a1), fmaxf(a2, a3));
#pragma unroll
      for (int off = 1; off < 16; off <<= 1) mx = fmaxf(mx, __shfl_xor(mx, off, 32));
      float nm = fmaxf(rowm[r], mx);
      float corr = __expf(rowm[r] - nm);
      rowm[r] = nm;
      float p0 = __expf(a0 - nm), p1 = __expf(a1 - nm);
      float p2 = __expf(a2 - nm), p3 = __expf(a3 - nm);
      float ps = (p0 + p1) + (p2 + p3);
#pragma unroll
      for (int off = 1; off < 16; off <<= 1) ps += __shfl_xor(ps, off, 32);
      rowl[r] = rowl[r] * corr + ps;
#pragma unroll
      for (int tt = 0; tt < 4; ++tt) acc[tt][r] *= corr;
      pS[wid][r + hl * 8][lanelo]      = f2bf(p0);
      pS[wid][r + hl * 8][lanelo + 16] = f2bf(p1);
      pS[wid][r + hl * 8][lanelo + 32] = f2bf(p2);
      pS[wid][r + hl * 8][lanelo + 48] = f2bf(p3);
    }
    // intra-wave LDS RAW: wait for P stores before re-reading as A-fragment
    asm volatile("s_wait_dscnt 0x0" ::: "memory");

    // O[16,64] += P[16,64] @ V[64,64]   (4 d tiles x 2 WMMAs)
    v16bf pa0 = ld16(&pS[wid][lanelo][hl * 8],      &pS[wid][lanelo][hl * 8 + 16]);
    v16bf pa1 = ld16(&pS[wid][lanelo][32 + hl * 8], &pS[wid][lanelo][32 + hl * 8 + 16]);
#pragma unroll
    for (int tt = 0; tt < 4; ++tt) {
      int dcol = tt * 16 + lanelo;
      v16bf bF0 = ld16(&vS[cur][dcol][hl * 8],      &vS[cur][dcol][hl * 8 + 16]);
      v16bf bF1 = ld16(&vS[cur][dcol][32 + hl * 8], &vS[cur][dcol][32 + hl * 8 + 16]);
      acc[tt] = wmma_bf16(pa0, bF0, acc[tt]);
      acc[tt] = wmma_bf16(pa1, bF1, acc[tt]);
    }
  }

  // finalize: divide by row sums, write ctx in [B,S,H,D_K] bf16
  float invl[8];
#pragma unroll
  for (int r = 0; r < 8; ++r) invl[r] = 1.0f / rowl[r];
#pragma unroll
  for (int tt = 0; tt < 4; ++tt) {
    int d = tt * 16 + lanelo;
#pragma unroll
    for (int r = 0; r < 8; ++r) {
      int row = qbase + r + hl * 8;
      ctx[(((size_t)b * SS + row) * NUM_HEADS + h) * D_K + d] = f2bf(acc[tt][r] * invl[r]);
    }
  }
}

// ---------------------------------------------------------------------------
// Kernel 4: RMSNorm over D_MODEL, one block per row
// ---------------------------------------------------------------------------
__global__ void rmsnorm(const float* __restrict__ in, const float* __restrict__ gamma,
                        float* __restrict__ out) {
  __shared__ float red[256];
  int row = blockIdx.x;
  const float* x = in + (size_t)row * D_MODEL;
  float s = 0.f;
  for (int i = threadIdx.x; i < D_MODEL; i += 256) { float v = x[i]; s += v * v; }
  red[threadIdx.x] = s;
  __syncthreads();
  for (int off = 128; off > 0; off >>= 1) {
    if ((int)threadIdx.x < off) red[threadIdx.x] += red[threadIdx.x + off];
    __syncthreads();
  }
  float rms = rsqrtf(red[0] / (float)D_MODEL + 1e-6f);
  for (int i = threadIdx.x; i < D_MODEL; i += 256)
    out[(size_t)row * D_MODEL + i] = x[i] * rms * gamma[i];
}

// ---------------------------------------------------------------------------
// Launch
// ---------------------------------------------------------------------------
extern "C" void kernel_launch(void* const* d_in, const int* in_sizes, int n_in,
                              void* d_out, int out_size, void* d_ws, size_t ws_size,
                              hipStream_t stream) {
  (void)in_sizes; (void)n_in; (void)out_size; (void)ws_size;

  const float* Q    = (const float*)d_in[0];
  const float* K    = (const float*)d_in[1];
  const float* V    = (const float*)d_in[2];
  const int*   mask = (const int*)  d_in[3];
  const float* wq = (const float*)d_in[4];  const float* bq = (const float*)d_in[5];
  const float* wk = (const float*)d_in[6];  const float* bk = (const float*)d_in[7];
  const float* wv = (const float*)d_in[8];  const float* bv = (const float*)d_in[9];
  const float* wo = (const float*)d_in[10]; const float* bo = (const float*)d_in[11];
  const float* wnq = (const float*)d_in[12]; const float* bnq = (const float*)d_in[13];
  const float* wnk = (const float*)d_in[14]; const float* bnk = (const float*)d_in[15];
  const float* temperature = (const float*)d_in[16];
  const float* gamma = (const float*)d_in[17];
  float* out = (float*)d_out;

  const int nBSD = BB * SS * D_MODEL;          // 4,194,304
  const int nW   = D_MODEL * D_MODEL;          // 1,048,576
  const int nN   = D_K * D_K;                  // 4,096
  const int nMask = SS * SS;                   // 4,194,304

  char* w = (char*)d_ws; size_t off = 0;
  auto carve = [&](size_t bytes) { void* p = w + off; off = (off + bytes + 255) & ~(size_t)255; return p; };
  __bf16* Qb   = (__bf16*)carve((size_t)nBSD * 2);
  __bf16* Kb   = (__bf16*)carve((size_t)nBSD * 2);
  __bf16* Vb   = (__bf16*)carve((size_t)nBSD * 2);
  __bf16* qp   = (__bf16*)carve((size_t)nBSD * 2);
  __bf16* kp   = (__bf16*)carve((size_t)nBSD * 2);
  __bf16* vtb  = (__bf16*)carve((size_t)nBSD * 2);   // [B,H,D_K,S]
  __bf16* qfb  = (__bf16*)carve((size_t)nBSD * 2);
  __bf16* kfb  = (__bf16*)carve((size_t)nBSD * 2);
  __bf16* ctxb = (__bf16*)carve((size_t)nBSD * 2);
  __bf16* wqT  = (__bf16*)carve((size_t)nW * 2);     // weights transposed [N,K]
  __bf16* wkT  = (__bf16*)carve((size_t)nW * 2);
  __bf16* wvT  = (__bf16*)carve((size_t)nW * 2);
  __bf16* woT  = (__bf16*)carve((size_t)nW * 2);
  __bf16* wnqT = (__bf16*)carve((size_t)nN * 2);
  __bf16* wnkT = (__bf16*)carve((size_t)nN * 2);
  unsigned* mb = (unsigned*)carve((size_t)(nMask / 32) * 4);
  float*  outf = (float*)carve((size_t)nBSD * 4);

  auto cvt = [&](const float* src, __bf16* dst, int n) {
    cvt_f32_bf16<<<(n / 4 + 255) / 256, 256, 0, stream>>>(src, dst, n);
  };
  auto cvtT = [&](const float* src, __bf16* dst, int Kd, int Nd) {
    cvt_transpose<<<dim3(Nd / 32, Kd / 32), dim3(32, 8), 0, stream>>>(src, dst, Kd, Nd);
  };
  cvt(Q, Qb, nBSD); cvt(K, Kb, nBSD); cvt(V, Vb, nBSD);
  cvtT(wq, wqT, D_MODEL, D_MODEL); cvtT(wk, wkT, D_MODEL, D_MODEL);
  cvtT(wv, wvT, D_MODEL, D_MODEL); cvtT(wo, woT, D_MODEL, D_MODEL);
  cvtT(wnq, wnqT, D_K, D_K);       cvtT(wnk, wnkT, D_K, D_K);
  mask_pack<<<nMask / 256, 256, 0, stream>>>(mask, mb);

  const int MBS = BB * SS;  // 4096
  dim3 blk(256);
  // QKV projections: [4096,1024] @ [1024,1024] + bias
  dim3 gProj(D_MODEL / 128, MBS / 128);
  gemm_bf16<<<gProj, blk, 0, stream>>>(Qb, wqT, bq, qp,  MBS, D_MODEL, D_MODEL, 0);
  gemm_bf16<<<gProj, blk, 0, stream>>>(Kb, wkT, bk, kp,  MBS, D_MODEL, D_MODEL, 0);
  gemm_bf16<<<gProj, blk, 0, stream>>>(Vb, wvT, bv, vtb, MBS, D_MODEL, D_MODEL, 3);

  // per-head feature maps (wnq shared across heads): [B*S*H,64] @ [64,64], tanh
  const int MH = MBS * NUM_HEADS;  // 65536
  dim3 gFeat(1, MH / 128);
  gemm_bf16<<<gFeat, blk, 0, stream>>>(qp, wnqT, bnq, qfb, MH, D_K, D_K, 1);
  gemm_bf16<<<gFeat, blk, 0, stream>>>(kp, wnkT, bnk, kfb, MH, D_K, D_K, 1);

  // flash attention
  dim3 gAtt(SS / 128, NUM_HEADS, BB);
  flash_attn<<<gAtt, dim3(256), 0, stream>>>(qfb, kfb, vtb, mb, temperature, ctxb);

  // output projection -> f32
  gemm_bf16<<<gProj, blk, 0, stream>>>(ctxb, woT, bo, outf, MBS, D_MODEL, D_MODEL, 2);

  // RMSNorm -> d_out
  rmsnorm<<<dim3(MBS), dim3(256), 0, stream>>>(outf, gamma, out);
}
